// TreeRNN_85761906966779
// MI455X (gfx1250) — matrix operand order repaired and checked by reading, not compiled, not run on hardware
//
#include <hip/hip_runtime.h>
#include <hip/hip_bf16.h>
#include <math.h>

// ---------------------------------------------------------------------------
// TreeRNN on MI455X (gfx1250, wave32, WMMA)
//   states carried in f16, GEMM via v_wmma_f32_16x16x32_f16, f32 accumulate.
//   Big levels (n >= 512): one wave per 16-row tile, streaming from HBM.
//   Tail levels (n = 256..1): fused into ONE workgroup with states in LDS,
//   workgroup barriers between levels (removes 9 latency-bound launches).
// ---------------------------------------------------------------------------

typedef _Float16 f16;
typedef __attribute__((ext_vector_type(16))) _Float16 v16h;
typedef __attribute__((ext_vector_type(8)))  _Float16 v8h;
typedef __attribute__((ext_vector_type(8)))  float    v8f;

#define TR_HDIM   128          // state width
#define TR_K2     256          // 2*HDIM, GEMM K
#define TR_NLEAF  524288

// ---------------------------------------------------------------------------
// Leaf embedding: states0[i,0:64]=Wpos[pos_idx[i]], states0[i,64:128]=Wwrd[wrd_idx[i]]
// ---------------------------------------------------------------------------
__global__ void tr_leaf_embed(const int* __restrict__ pos_idx,
                              const int* __restrict__ wrd_idx,
                              const float* __restrict__ Wpos,   // (64, 64)
                              const float* __restrict__ Wwrd,   // (100000, 64)
                              f16* __restrict__ out)            // (NLEAF, 128) f16
{
    int tid = blockIdx.x * blockDim.x + threadIdx.x;
    if (tid >= TR_NLEAF * TR_HDIM) return;
    int leaf = tid >> 7;
    int j    = tid & 127;
    float v = (j < 64) ? Wpos[(size_t)pos_idx[leaf] * 64 + j]
                       : Wwrd[(size_t)wrd_idx[leaf] * 64 + (j - 64)];
    out[tid] = (f16)v;
}

// ---------------------------------------------------------------------------
// Convert Wc_w (128 x 256, f32, row-major) to f16, same layout.
// out[m][col] = sum_k lr[m][k] * Wc_w[col][k]  -> Wc_w IS B-transposed (NxK).
// ---------------------------------------------------------------------------
__global__ void tr_cvt_w(const float* __restrict__ w, f16* __restrict__ wh, int total)
{
    int tid = blockIdx.x * blockDim.x + threadIdx.x;
    if (tid < total) wh[tid] = (f16)w[tid];
}

// ---------------------------------------------------------------------------
// Big level:  out[m,:] = tanh(Wnon[nonIdx[m],:] + A[m,:] @ Wc_w.T + b)
// One wave per 16-row tile; 8 N-tiles x 8 K-chunks = 64 WMMA.
// ---------------------------------------------------------------------------
__global__ void __launch_bounds__(32)
tr_level(const f16*  __restrict__ A,        // (n, 256) f16
         const f16*  __restrict__ BT,       // (128, 256) f16 (= Wc_w)
         const float* __restrict__ Wnon,    // (64, 128)
         const float* __restrict__ bias,    // (128,)
         const int*   __restrict__ nonIdx,  // already offset
         f16*         __restrict__ outH,    // (n, 128) f16
         int n)
{
    const int lane  = threadIdx.x;
    const int half  = lane >> 4;
    const int l15   = lane & 15;
    const int mBase = blockIdx.x << 4;

    // Speculative prefetch of next M-tile's A rows (global_prefetch_b8).
    __builtin_prefetch(A + (size_t)(mBase + 16 + l15) * TR_K2, 0, 3);

    const int  arow  = mBase + l15;
    const f16* aRowP = A + (size_t)arow * TR_K2 + half * 16;

    v8f acc[8] = {};
    #pragma unroll
    for (int kc = 0; kc < 8; ++kc) {
        // A fragment (16x32 f16, contiguous K, matching B):
        //  lane<16 : halves 0..15 = K+0..15
        //  lane>=16: halves 0..15 = K+16..31
        const v8h lo = *(const v8h*)(aRowP + kc * 32);
        const v8h hi = *(const v8h*)(aRowP + kc * 32 + 8);
        const v16h a = __builtin_shufflevector(lo, hi,
                                               0, 1, 2, 3, 4, 5, 6, 7,
                                               8, 9, 10, 11, 12, 13, 14, 15);
        #pragma unroll
        for (int nt = 0; nt < 8; ++nt) {
            // B fragment (32x16 f16): lane = column, 16 contiguous K halves.
            const v16h b = *(const v16h*)(BT + (size_t)(nt * 16 + l15) * TR_K2
                                             + kc * 32 + half * 16);
            acc[nt] = __builtin_amdgcn_wmma_f32_16x16x32_f16(
                          false, a, false, b, (short)0, acc[nt], false, false);
        }
    }

    // Epilogue: C layout -> row = mBase + r + 8*half, col = nt*16 + l15.
    int nrow[8];
    #pragma unroll
    for (int r = 0; r < 8; ++r) nrow[r] = nonIdx[mBase + r + half * 8];

    #pragma unroll
    for (int nt = 0; nt < 8; ++nt) {
        const int   col  = nt * 16 + l15;
        const float bcol = bias[col];
        #pragma unroll
        for (int r = 0; r < 8; ++r) {
            const int row = mBase + r + half * 8;
            float v = acc[nt][r] + Wnon[(size_t)nrow[r] * TR_HDIM + col] + bcol;
            outH[(size_t)row * TR_HDIM + col] = (f16)tanhf(v);
        }
    }
}

// ---------------------------------------------------------------------------
// Fused tail: levels n = 256, 128, ..., 1 in ONE workgroup (16 waves).
// States live in LDS (192 KB of the WGP's 320 KB); A fragments come from LDS
// via ds_load_b128; workgroup barriers between levels.
//   Ain : (512, 128) f16 global (output of the n=512 level)
//   outF: final 128 f32 result
// ---------------------------------------------------------------------------
__global__ void __launch_bounds__(512)
tr_tail(const f16*  __restrict__ Ain,
        const f16*  __restrict__ BT,
        const float* __restrict__ Wnon,
        const float* __restrict__ bias,
        const int*   __restrict__ nonIdx,   // offset to start of n=256 level
        float*       __restrict__ outF)
{
    extern __shared__ f16 lds[];
    f16* s0 = lds;                          // capacity 512 x 128
    f16* s1 = lds + 512 * TR_HDIM;          // capacity 256 x 128

    const int tid = threadIdx.x;

    // Stage the 512x128 f16 input (128 KB) into LDS: 8192 x 16B.
    {
        const uint4* src = (const uint4*)Ain;
        uint4*       dst = (uint4*)s0;
        for (int i = tid; i < 8192; i += 512) dst[i] = src[i];
    }
    __syncthreads();

    const int lane   = tid & 31;
    const int waveId = tid >> 5;
    const int half   = lane >> 4;
    const int l15    = lane & 15;

    f16* sIn  = s0;
    f16* sOut = s1;
    int  off  = 0;

    for (int n = 256; n >= 1; n >>= 1) {
        const int mTiles = (n + 15) >> 4;
        if (waveId < mTiles) {                       // wave-uniform: EXEC stays full
            const int  mBase  = waveId << 4;
            const int  arow   = mBase + l15;
            const bool aValid = (arow < n);
            const f16* aRowP  = sIn + arow * TR_K2 + half * 16;

            v8f acc[8] = {};
            #pragma unroll
            for (int kc = 0; kc < 8; ++kc) {
                v16h a = {};
                if (aValid) {                        // reconverges before WMMA
                    const v8h lo = *(const v8h*)(aRowP + kc * 32);
                    const v8h hi = *(const v8h*)(aRowP + kc * 32 + 8);
                    a = __builtin_shufflevector(lo, hi,
                                                0, 1, 2, 3, 4, 5, 6, 7,
                                                8, 9, 10, 11, 12, 13, 14, 15);
                }
                #pragma unroll
                for (int nt = 0; nt < 8; ++nt) {
                    const v16h b = *(const v16h*)(BT + (size_t)(nt * 16 + l15) * TR_K2
                                                     + kc * 32 + half * 16);
                    acc[nt] = __builtin_amdgcn_wmma_f32_16x16x32_f16(
                                  false, a, false, b, (short)0, acc[nt], false, false);
                }
            }

            int nrow[8];
            #pragma unroll
            for (int r = 0; r < 8; ++r) {
                const int row = mBase + r + half * 8;
                nrow[r] = (row < n) ? nonIdx[off + row] : 0;
            }
            #pragma unroll
            for (int nt = 0; nt < 8; ++nt) {
                const int   col  = nt * 16 + l15;
                const float bcol = bias[col];
                #pragma unroll
                for (int r = 0; r < 8; ++r) {
                    const int row = mBase + r + half * 8;
                    if (row < n) {
                        float v = acc[nt][r] + Wnon[(size_t)nrow[r] * TR_HDIM + col] + bcol;
                        float t = tanhf(v);
                        sOut[row * TR_HDIM + col] = (f16)t;
                        if (n == 1) outF[col] = t;   // only row 0 at final level
                    }
                }
            }
        }
        __syncthreads();
        off += n;
        f16* t = sIn; sIn = sOut; sOut = t;
    }
}

// ---------------------------------------------------------------------------
// Host-side launch. Inputs (setup_inputs order):
//   0 pos_idx (i32)  1 wrd_idx (i32)  2 non_idx (i32)
//   3 Wwrd (f32)     4 Wpos (f32)     5 Wnon (f32)    6 Wc_w (f32)  7 Wc_b (f32)
// Output: 128 f32.
// Workspace (~192.1 MiB): bufA (128 MiB f16) | bufB (64 MiB f16) | Wc_w f16 (64 KiB)
// ---------------------------------------------------------------------------
extern "C" void kernel_launch(void* const* d_in, const int* in_sizes, int n_in,
                              void* d_out, int out_size, void* d_ws, size_t ws_size,
                              hipStream_t stream)
{
    const int*   pos_idx = (const int*)  d_in[0];
    const int*   wrd_idx = (const int*)  d_in[1];
    const int*   non_idx = (const int*)  d_in[2];
    const float* Wwrd    = (const float*)d_in[3];
    const float* Wpos    = (const float*)d_in[4];
    const float* Wnon    = (const float*)d_in[5];
    const float* Wc_w    = (const float*)d_in[6];
    const float* Wc_b    = (const float*)d_in[7];

    char* ws = (char*)d_ws;
    const size_t szA = (size_t)TR_NLEAF       * TR_HDIM * sizeof(f16);
    const size_t szB = (size_t)(TR_NLEAF / 2) * TR_HDIM * sizeof(f16);
    f16* bufA = (f16*)(ws);
    f16* bufB = (f16*)(ws + szA);
    f16* WcH  = (f16*)(ws + szA + szB);

    // 1) leaf embedding -> f16 states
    {
        const int total = TR_NLEAF * TR_HDIM;
        tr_leaf_embed<<<(total + 255) / 256, 256, 0, stream>>>(
            pos_idx, wrd_idx, Wpos, Wwrd, bufA);
    }
    // 2) weight conversion (tiny)
    {
        const int total = TR_HDIM * TR_K2;
        tr_cvt_w<<<(total + 255) / 256, 256, 0, stream>>>(Wc_w, WcH, total);
    }

    // 3) big levels n = 262144 .. 512 (all tiles full: n multiple of 16)
    long offset = 0;
    int  n      = TR_NLEAF / 2;
    f16* cur = bufA;
    f16* nxt = bufB;
    while (n >= 512) {
        tr_level<<<n / 16, 32, 0, stream>>>(
            cur, WcH, Wnon, Wc_b, non_idx + offset, nxt, n);
        offset += n;
        f16* t = cur; cur = nxt; nxt = t;
        n >>= 1;
    }

    // 4) fused tail n = 256 .. 1 in one workgroup; states in LDS.
    //    cur now holds the 512x128 output of the n=512 level.
    const size_t ldsBytes = (size_t)(512 + 256) * TR_HDIM * sizeof(f16); // 192 KiB
    tr_tail<<<1, 512, ldsBytes, stream>>>(
        cur, WcH, Wnon, Wc_b, non_idx + offset, (float*)d_out);
}
